// DFT_29789893165287
// MI455X (gfx1250) — compile-verified, matrix operand order checked by reading, not run on hardware
//
#include <hip/hip_runtime.h>

// ---------------------------------------------------------------------------
// Batched 256-point DFT as 4 real GEMMs on CDNA5 WMMA (f16 in, f32 accum).
//   real = x_r*W_r - x_i*W_i ;  imag = x_r*W_i + x_i*W_r ;  * (1/16)
// x: [B=262144, 256] fp32 (row-major), W: [256,256] fp32, out: fp32.
// HBM-bound (~1.07 GB => ~46us @ 23.3 TB/s); f16 WMMA compute ~27us.
// ---------------------------------------------------------------------------

typedef __attribute__((ext_vector_type(16))) _Float16 v16h;
typedef __attribute__((ext_vector_type(8)))  _Float16 v8h;
typedef __attribute__((ext_vector_type(8)))  float    v8f;

#define WMMA_F16(A, B, C) \
  __builtin_amdgcn_wmma_f32_16x16x32_f16(false, (A), false, (B), (short)0, (C), false, false)

constexpr int kN       = 256;   // DFT size (K and N of the GEMM)
constexpr int kBlkRows = 256;   // x rows per workgroup (8 waves x 32 rows)
constexpr int kBlkCols = 32;    // output cols per workgroup
constexpr int kLdsK    = 264;   // padded K stride (+8 halves => conflict-free b128 reads)

// A-fragment (16x32 f16, ISA layout): elements 0..7 = K[kb+8h .. +7],
// elements 8..15 = K[kb+16+8h .. +7], built from fp32 global with on-the-fly cvt.
__device__ __forceinline__ v16h frag_from_f32(const float* __restrict__ p, int h) {
  const float4* q0 = reinterpret_cast<const float4*>(p + 8 * h);
  const float4* q1 = reinterpret_cast<const float4*>(p + 16 + 8 * h);
  const float4 a0 = q0[0], a1 = q0[1];
  const float4 b0 = q1[0], b1 = q1[1];
  v16h r;
  r[0]  = (_Float16)a0.x; r[1]  = (_Float16)a0.y; r[2]  = (_Float16)a0.z; r[3]  = (_Float16)a0.w;
  r[4]  = (_Float16)a1.x; r[5]  = (_Float16)a1.y; r[6]  = (_Float16)a1.z; r[7]  = (_Float16)a1.w;
  r[8]  = (_Float16)b0.x; r[9]  = (_Float16)b0.y; r[10] = (_Float16)b0.z; r[11] = (_Float16)b0.w;
  r[12] = (_Float16)b1.x; r[13] = (_Float16)b1.y; r[14] = (_Float16)b1.z; r[15] = (_Float16)b1.w;
  return r;
}

// B-fragment (32x16 f16): W stored transposed [n][k] in LDS, so each lane's two
// 8-element K-runs are contiguous 16-byte chunks => 2x ds_read_b128.
__device__ __forceinline__ v16h frag_from_lds(const _Float16* p, int h) {
  const v8h lo = *reinterpret_cast<const v8h*>(p + 8 * h);
  const v8h hi = *reinterpret_cast<const v8h*>(p + 16 + 8 * h);
  return __builtin_shufflevector(lo, hi, 0,1,2,3,4,5,6,7,8,9,10,11,12,13,14,15);
}

__global__ void __launch_bounds__(256)
dft256_wmma(const float* __restrict__ xr, const float* __restrict__ xi,
            const float* __restrict__ Wr, const float* __restrict__ Wi,
            float* __restrict__ outr, float* __restrict__ outi)
{
  __shared__ _Float16 ldsW[2 * kBlkCols * kLdsK];   // 33,792 B

  const int tid = threadIdx.x;
  const int cb  = blockIdx.x * kBlkCols;
  const size_t rb = (size_t)blockIdx.y * kBlkRows;

  // ---- Preload W slices: fp32 global -> f16, transposed [n][k] into LDS ----
  {
    const int f4   = tid & 7;    // which float4 within a 32-col row (coalesced)
    const int krow = tid >> 3;   // 32 k-rows per sweep
    #pragma unroll
    for (int mat = 0; mat < 2; ++mat) {
      const float* src = mat ? Wi : Wr;
      _Float16* dst = ldsW + mat * (kBlkCols * kLdsK);
      #pragma unroll
      for (int it = 0; it < kN / 32; ++it) {
        const int k = it * 32 + krow;
        const float4 v = *reinterpret_cast<const float4*>(src + (size_t)k * kN + cb + f4 * 4);
        const int n = f4 * 4;
        dst[(n + 0) * kLdsK + k] = (_Float16)v.x;
        dst[(n + 1) * kLdsK + k] = (_Float16)v.y;
        dst[(n + 2) * kLdsK + k] = (_Float16)v.z;
        dst[(n + 3) * kLdsK + k] = (_Float16)v.w;
      }
    }
  }
  __syncthreads();

  const int lane = tid & 31;
  const int wave = tid >> 5;     // 8 waves = 8 row groups of 32 rows
  const int lm   = lane & 15;    // A: row-in-tile / B,C,D: col-in-tile
  const int h    = lane >> 4;    // K-half select (A/B), M+8 select (C/D)

  const size_t row0 = rb + (size_t)wave * 32 + lm;
  const float* pr0 = xr + row0 * kN;          // mtile 0
  const float* pr1 = pr0 + 16 * kN;           // mtile 1
  const float* pi0 = xi + row0 * kN;
  const float* pi1 = pi0 + 16 * kN;

  const _Float16* br0 = ldsW + lm * kLdsK;          // W_real, ntile 0
  const _Float16* br1 = br0 + 16 * kLdsK;           // W_real, ntile 1
  const _Float16* bi0 = br0 + kBlkCols * kLdsK;     // W_imag, ntile 0
  const _Float16* bi1 = bi0 + 16 * kLdsK;

  v8f re00 = {}, re01 = {}, re10 = {}, re11 = {};
  v8f im00 = {}, im01 = {}, im10 = {}, im11 = {};

  #pragma unroll 2
  for (int kb = 0; kb < kN; kb += 32) {
    const v16h aR0  = frag_from_f32(pr0 + kb, h);
    const v16h aR1  = frag_from_f32(pr1 + kb, h);
    const v16h aI0  = frag_from_f32(pi0 + kb, h);
    const v16h aI1  = frag_from_f32(pi1 + kb, h);
    const v16h aIn0 = -aI0;   // ISA: f16 WMMA NEG[1:0] must be 0, negate in VALU
    const v16h aIn1 = -aI1;
    const v16h bR0  = frag_from_lds(br0 + kb, h);
    const v16h bR1  = frag_from_lds(br1 + kb, h);
    const v16h bI0  = frag_from_lds(bi0 + kb, h);
    const v16h bI1  = frag_from_lds(bi1 + kb, h);

    // 16 WMMAs; each accumulator touched 8 apart (>5 insn WMMA->WMMA hazard gap)
    re00 = WMMA_F16(aR0,  bR0, re00);
    re01 = WMMA_F16(aR0,  bR1, re01);
    re10 = WMMA_F16(aR1,  bR0, re10);
    re11 = WMMA_F16(aR1,  bR1, re11);
    im00 = WMMA_F16(aR0,  bI0, im00);
    im01 = WMMA_F16(aR0,  bI1, im01);
    im10 = WMMA_F16(aR1,  bI0, im10);
    im11 = WMMA_F16(aR1,  bI1, im11);
    re00 = WMMA_F16(aIn0, bI0, re00);
    re01 = WMMA_F16(aIn0, bI1, re01);
    re10 = WMMA_F16(aIn1, bI0, re10);
    re11 = WMMA_F16(aIn1, bI1, re11);
    im00 = WMMA_F16(aI0,  bR0, im00);
    im01 = WMMA_F16(aI0,  bR1, im01);
    im10 = WMMA_F16(aI1,  bR0, im10);
    im11 = WMMA_F16(aI1,  bR1, im11);
  }

  // ---- Epilogue: scale by 1/sqrt(256)=1/16, store per C/D VGPR layout ----
  constexpr float scale = 0.0625f;
  const size_t colBase = (size_t)cb + lm;
  const size_t rBase   = rb + (size_t)wave * 32 + (size_t)h * 8;

  auto store_tile = [&](float* __restrict__ out, const v8f& acc, int mt, int nt) {
    float* o = out + (rBase + (size_t)mt * 16) * kN + colBase + (size_t)nt * 16;
    #pragma unroll
    for (int v = 0; v < 8; ++v)
      o[(size_t)v * kN] = acc[v] * scale;
  };

  store_tile(outr, re00, 0, 0); store_tile(outr, re01, 0, 1);
  store_tile(outr, re10, 1, 0); store_tile(outr, re11, 1, 1);
  store_tile(outi, im00, 0, 0); store_tile(outi, im01, 0, 1);
  store_tile(outi, im10, 1, 0); store_tile(outi, im11, 1, 1);
}

extern "C" void kernel_launch(void* const* d_in, const int* in_sizes, int n_in,
                              void* d_out, int out_size, void* d_ws, size_t ws_size,
                              hipStream_t stream) {
  const float* xr = (const float*)d_in[0];
  const float* xi = (const float*)d_in[1];
  const float* Wr = (const float*)d_in[2];
  const float* Wi = (const float*)d_in[3];
  float* out = (float*)d_out;

  const int rows = in_sizes[0] / kN;                 // 262144
  float* outr = out;
  float* outi = out + (size_t)rows * kN;

  dim3 grid(kN / kBlkCols, rows / kBlkRows);         // (8, 1024); col-block fastest
  dim3 block(256);                                   // 8 wave32
  dft256_wmma<<<grid, block, 0, stream>>>(xr, xi, Wr, Wi, outr, outi);
}
